// SparseMixtureAttention_74577812127918
// MI455X (gfx1250) — matrix operand order repaired and checked by reading, not compiled, not run on hardware
//
#include <hip/hip_runtime.h>
#include <hip/hip_bf16.h>

// ---- CDNA5 wave32 WMMA types -------------------------------------------------
typedef __attribute__((ext_vector_type(16))) __bf16        v16bf;
typedef __attribute__((ext_vector_type(8)))  float         v8f;
typedef __attribute__((ext_vector_type(8)))  unsigned int  v8u;

#define SEQ   2048
#define NH    16
#define HD    64
#define EMB   1024
#define NB    2
#define MROWS (NB * SEQ)   // 4096

// round-to-nearest-even fp32 -> bf16 bits
__device__ __forceinline__ unsigned short f2bf(float f) {
  unsigned u = __builtin_bit_cast(unsigned, f);
  unsigned r = u + 0x7FFFu + ((u >> 16) & 1u);
  return (unsigned short)(r >> 16);
}

__device__ __forceinline__ v8f wmma_bf16(v8u a, v8u b, v8f c) {
  return __builtin_amdgcn_wmma_f32_16x16x32_bf16(
      false, __builtin_bit_cast(v16bf, a),
      false, __builtin_bit_cast(v16bf, b),
      (short)0, c, false, false);
}

// reductions within each 16-lane half of a wave32 (C-fragment row groups)
__device__ __forceinline__ float rmax16(float v) {
#pragma unroll
  for (int m = 1; m < 16; m <<= 1) v = fmaxf(v, __shfl_xor(v, m));
  return v;
}
__device__ __forceinline__ float rsum16(float v) {
#pragma unroll
  for (int m = 1; m < 16; m <<= 1) v += __shfl_xor(v, m);
  return v;
}

// ---- fp32 -> bf16 cast kernel -----------------------------------------------
__global__ void cvt_f32_bf16(const float* __restrict__ in,
                             unsigned short* __restrict__ out, int n) {
  for (int i = blockIdx.x * blockDim.x + threadIdx.x; i < n;
       i += gridDim.x * blockDim.x)
    out[i] = f2bf(in[i]);
}

// ---- software-pipelined 16(M)x64(N) WMMA tile, K = EMB ----------------------
// A: [M][K] bf16 row-major (K-pairs contiguous).  B: [N][K] bf16 (W natural
// layout, so B(k,n)=W[n][k] with K-pairs contiguous).
// Copy-free double buffering: unroll k by 2 so buffer sets alternate roles.
__device__ __forceinline__ void gemm_tile_16x64(
    const unsigned short* __restrict__ A, const unsigned short* __restrict__ Bw,
    int row0, int n0, int lo16, int kh, v8f (&acc)[4]) {
  const unsigned short* ap = A + ((size_t)(row0 + lo16) << 10);

  auto loadA = [&](int k0, v8u& au) {
#pragma unroll
    for (int i = 0; i < 8; ++i) {
      int kk = k0 + ((i < 4) ? 0 : 16) + kh * 8 + (i & 3) * 2;
      au[i] = *(const unsigned int*)(ap + kk);
    }
  };
  auto loadB = [&](int k0, int f, v8u& bu) {
    const unsigned short* bp = Bw + ((size_t)(n0 + f * 16 + lo16) << 10);
#pragma unroll
    for (int i = 0; i < 8; ++i)
      bu[i] = *(const unsigned int*)(bp + k0 + ((i < 4) ? 0 : 16) + kh * 8 + (i & 3) * 2);
  };

  v8u a0, b0[4], a1, b1[4];
  loadA(0, a0);
#pragma unroll
  for (int f = 0; f < 4; ++f) loadB(0, f, b0[f]);

#pragma unroll 1
  for (int k0 = 0; k0 < EMB - 64; k0 += 64) {
    // stage 1: load set-1 for k0+32, consume set-0 (k0)
    loadA(k0 + 32, a1);
#pragma unroll
    for (int f = 0; f < 4; ++f) loadB(k0 + 32, f, b1[f]);
#pragma unroll
    for (int f = 0; f < 4; ++f) acc[f] = wmma_bf16(a0, b0[f], acc[f]);
    // stage 2: load set-0 for k0+64, consume set-1 (k0+32)
    loadA(k0 + 64, a0);
#pragma unroll
    for (int f = 0; f < 4; ++f) loadB(k0 + 64, f, b0[f]);
#pragma unroll
    for (int f = 0; f < 4; ++f) acc[f] = wmma_bf16(a1, b1[f], acc[f]);
  }
  // tail: k0 = EMB-64 in set-0, EMB-32 in set-1
  loadA(EMB - 32, a1);
#pragma unroll
  for (int f = 0; f < 4; ++f) loadB(EMB - 32, f, b1[f]);
#pragma unroll
  for (int f = 0; f < 4; ++f) acc[f] = wmma_bf16(a0, b0[f], acc[f]);
#pragma unroll
  for (int f = 0; f < 4; ++f) acc[f] = wmma_bf16(a1, b1[f], acc[f]);
}

// ---- QKV GEMM: [4096,1024] x [1024,3072] + bias, scatter to q/k/vT ----------
__global__ __launch_bounds__(256) void qkv_gemm(
    const unsigned short* __restrict__ xb,
    const unsigned short* __restrict__ wab,
    const float* __restrict__ bias,
    unsigned short* __restrict__ qb,
    unsigned short* __restrict__ kbuf,
    unsigned short* __restrict__ vtb) {
  const int wid  = threadIdx.x >> 5;
  const int lane = threadIdx.x & 31;
  const int tile = blockIdx.x * 8 + wid;     // 12288 tiles = 256 x 48
  const int mt   = tile & 255;
  const int nt   = tile >> 8;                // 0..47
  const int row0 = mt << 4;
  const int n0   = nt << 6;
  const int lo16 = lane & 15;
  const int kh   = lane >> 4;

  v8f acc[4] = {};
  gemm_tile_16x64(xb, wab, row0, n0, lo16, kh, acc);

  // epilogue: split q/k/v, reshape heads, transpose v
#pragma unroll
  for (int f = 0; f < 4; ++f) {
    int n = n0 + f * 16 + lo16;
    int which = n >> 10;
    int hn = n & 1023;
    int h = hn >> 6, d = hn & 63;
    float bv = bias[n];
#pragma unroll
    for (int v = 0; v < 8; ++v) {
      int m = row0 + v + 8 * kh;
      int b = m >> 11, t = m & 2047;
      int bh = b * NH + h;
      unsigned short s = f2bf(acc[f][v] + bv);
      if (which == 0)      qb  [((size_t)bh * SEQ + t) * HD + d] = s;
      else if (which == 1) kbuf[((size_t)bh * SEQ + t) * HD + d] = s;
      else                 vtb [((size_t)bh * HD  + d) * SEQ + t] = s;
    }
  }
}

// ---- sparse-masked flash attention: one wave per (b,h, 16-row tile) ---------
__global__ __launch_bounds__(128) void attn_kernel(
    const unsigned short* __restrict__ qb,
    const unsigned short* __restrict__ kbuf,
    const unsigned short* __restrict__ vtb,
    const float* __restrict__ mlog,
    unsigned short* __restrict__ yb) {
  __shared__ __align__(16) unsigned short ptile[4][16 * 32];
  const int wid  = threadIdx.x >> 5;
  const int lane = threadIdx.x & 31;
  const int tile = blockIdx.x * 4 + wid;     // 4096 tiles = 32 x 128
  const int bh   = tile >> 7;                // b*16 + h
  const int t0   = (tile & 127) << 4;
  const int h    = bh & 15;
  const int lo16 = lane & 15;
  const int kh   = lane >> 4;
  unsigned short* pt = ptile[wid];
  const float NEG_INF = -__builtin_inff();

  // Q A-fragments (16 rows x 64 K), held for whole tile
  v8u qa[2];
#pragma unroll
  for (int g = 0; g < 2; ++g)
#pragma unroll
    for (int i = 0; i < 8; ++i) {
      int kk = g * 32 + ((i < 4) ? 0 : 16) + kh * 8 + (i & 3) * 2;
      qa[g][i] =
          *(const unsigned int*)(qb + ((size_t)bh * SEQ + t0 + lo16) * HD + kk);
    }

  v8f o[4] = {};
  float mrun[8], lrun[8];
#pragma unroll
  for (int v = 0; v < 8; ++v) { mrun[v] = NEG_INF; lrun[v] = 0.0f; }

  const int send = t0 + 16;                  // causal: s <= t0+15
  for (int cs = 0; cs < send; cs += 32) {
    // ---- issue the HBM-bound mask loads first (overlap with score WMMAs)
    float lg[2][8];
#pragma unroll
    for (int half = 0; half < 2; ++half)
#pragma unroll
      for (int v = 0; v < 8; ++v) {
        int trow = t0 + v + 8 * kh;
        int s    = cs + half * 16 + lo16;
        lg[half][v] = mlog[((size_t)h * SEQ + trow) * SEQ + s];
      }
    if (cs + 32 < send)
      __builtin_prefetch(mlog + ((size_t)h * SEQ + t0 + lo16) * SEQ + cs + 32, 0, 0);

    // ---- scores S = Q K^T  (16 x 32, K=64)
    v8f sc[2] = {};
#pragma unroll
    for (int half = 0; half < 2; ++half) {
#pragma unroll
      for (int g = 0; g < 2; ++g) {
        v8u ku;
#pragma unroll
        for (int i = 0; i < 8; ++i) {
          int kd = g * 32 + ((i < 4) ? 0 : 16) + kh * 8 + (i & 3) * 2;
          int s  = cs + half * 16 + lo16;
          ku[i] =
              *(const unsigned int*)(kbuf + ((size_t)bh * SEQ + s) * HD + kd);
        }
        sc[half] = wmma_bf16(qa[g], ku, sc[half]);
      }
    }

    // ---- V B-fragments early: latency hides under the softmax VALU work
    v8u vu[4];
#pragma unroll
    for (int f = 0; f < 4; ++f)
#pragma unroll
      for (int i = 0; i < 8; ++i) {
        int ks = ((i < 4) ? 0 : 16) + kh * 8 + (i & 3) * 2;
        int d  = f * 16 + lo16;
        vu[f][i] =
            *(const unsigned int*)(vtb + ((size_t)bh * HD + d) * SEQ + cs + ks);
      }

    // ---- mask (sigmoid(logit)>0.5 <=> logit>0) & causal, online softmax
    float sval[2][8];
    unsigned vmask[2] = {0u, 0u};
    float mch[8];
#pragma unroll
    for (int v = 0; v < 8; ++v) mch[v] = NEG_INF;
#pragma unroll
    for (int half = 0; half < 2; ++half)
#pragma unroll
      for (int v = 0; v < 8; ++v) {
        int trow = t0 + v + 8 * kh;
        int s    = cs + half * 16 + lo16;
        float val = sc[half][v] * 0.125f;   // 1/sqrt(64)
        sval[half][v] = val;
        bool ok = (lg[half][v] > 0.0f) && (s <= trow);
        if (ok) { vmask[half] |= 1u << v; mch[v] = fmaxf(mch[v], val); }
      }
#pragma unroll
    for (int v = 0; v < 8; ++v) mch[v] = rmax16(mch[v]);

    float mnew[8], alpha[8], lsum[8];
#pragma unroll
    for (int v = 0; v < 8; ++v) {
      mnew[v]  = fmaxf(mrun[v], mch[v]);
      alpha[v] = (mrun[v] == NEG_INF) ? 0.0f : __expf(mrun[v] - mnew[v]);
      lsum[v]  = 0.0f;
    }
#pragma unroll
    for (int half = 0; half < 2; ++half)
#pragma unroll
      for (int v = 0; v < 8; ++v) {
        float p = ((vmask[half] >> v) & 1u)
                      ? __expf(sval[half][v] - mnew[v]) : 0.0f;
        lsum[v] += p;
        // C-layout -> LDS row-major [16][32]
        pt[(v + 8 * kh) * 32 + half * 16 + lo16] = f2bf(p);
      }
#pragma unroll
    for (int v = 0; v < 8; ++v) {
      lsum[v] = rsum16(lsum[v]);
      lrun[v] = lrun[v] * alpha[v] + lsum[v];
      mrun[v] = mnew[v];
    }
#pragma unroll
    for (int f = 0; f < 4; ++f)
#pragma unroll
      for (int v = 0; v < 8; ++v) o[f][v] *= alpha[v];

    // wait for this wave's cross-lane LDS stores, then read P in A-layout
    asm volatile("s_wait_dscnt 0" ::: "memory");
    v8u pu;
#pragma unroll
    for (int i = 0; i < 8; ++i) {
      int kk = ((i < 4) ? 0 : 16) + kh * 8 + (i & 3) * 2;
      pu[i] = *(const unsigned int*)(pt + lo16 * 32 + kk);
    }
    // O += P V
#pragma unroll
    for (int f = 0; f < 4; ++f) o[f] = wmma_bf16(pu, vu[f], o[f]);
  }

  // normalize, reassemble [B,T,H*D] as bf16 for the projection GEMM
  const int b = bh >> 4;
#pragma unroll
  for (int f = 0; f < 4; ++f) {
    int d = f * 16 + lo16;
#pragma unroll
    for (int v = 0; v < 8; ++v) {
      int trow = t0 + v + 8 * kh;
      float val = o[f][v] / lrun[v];
      yb[((size_t)(b * SEQ + trow)) * EMB + h * HD + d] = f2bf(val);
    }
  }
}

// ---- output projection: [4096,1024] x [1024,1024] + bias, fp32 out ----------
__global__ __launch_bounds__(256) void proj_gemm(
    const unsigned short* __restrict__ yb,
    const unsigned short* __restrict__ wpb,
    const float* __restrict__ bias,
    float* __restrict__ out) {
  const int wid  = threadIdx.x >> 5;
  const int lane = threadIdx.x & 31;
  const int tile = blockIdx.x * 8 + wid;     // 4096 tiles = 256 x 16
  const int mt   = tile & 255;
  const int nt   = tile >> 8;                // 0..15
  const int row0 = mt << 4;
  const int n0   = nt << 6;
  const int lo16 = lane & 15;
  const int kh   = lane >> 4;

  v8f acc[4] = {};
  gemm_tile_16x64(yb, wpb, row0, n0, lo16, kh, acc);

#pragma unroll
  for (int f = 0; f < 4; ++f) {
    int n = n0 + f * 16 + lo16;
    float bv = bias[n];
#pragma unroll
    for (int v = 0; v < 8; ++v) {
      int m = row0 + v + 8 * kh;
      out[(size_t)m * EMB + n] = acc[f][v] + bv;
    }
  }
}

// ---- host-side orchestration ------------------------------------------------
extern "C" void kernel_launch(void* const* d_in, const int* in_sizes, int n_in,
                              void* d_out, int out_size, void* d_ws,
                              size_t ws_size, hipStream_t stream) {
  const float* x    = (const float*)d_in[0];
  const float* mlog = (const float*)d_in[1];
  const float* Wa   = (const float*)d_in[2];
  const float* ba   = (const float*)d_in[3];
  const float* Wp   = (const float*)d_in[4];
  const float* bp   = (const float*)d_in[5];
  float* out = (float*)d_out;

  char* ws = (char*)d_ws;
  unsigned short* xb  = (unsigned short*)(ws);                        // 8 MB
  unsigned short* wab = (unsigned short*)(ws + ((size_t)8  << 20));   // 6 MB
  unsigned short* wpb = (unsigned short*)(ws + ((size_t)14 << 20));   // 2 MB
  unsigned short* qb  = (unsigned short*)(ws + ((size_t)16 << 20));   // 8 MB
  unsigned short* kb  = (unsigned short*)(ws + ((size_t)24 << 20));   // 8 MB
  unsigned short* vtb = (unsigned short*)(ws + ((size_t)32 << 20));   // 8 MB
  unsigned short* yb  = (unsigned short*)(ws + ((size_t)40 << 20));   // 8 MB

  cvt_f32_bf16<<<2048, 256, 0, stream>>>(x,  xb,  MROWS * EMB);
  cvt_f32_bf16<<<2048, 256, 0, stream>>>(Wa, wab, 3 * EMB * EMB);
  cvt_f32_bf16<<<1024, 256, 0, stream>>>(Wp, wpb, EMB * EMB);

  qkv_gemm<<<1536, 256, 0, stream>>>(xb, wab, ba, qb, kb, vtb);
  attn_kernel<<<1024, 128, 0, stream>>>(qb, kb, vtb, mlog, yb);
  proj_gemm<<<512, 256, 0, stream>>>(yb, wpb, bp, out);
}
